// MeshTransformerLayer_83373905150281
// MI455X (gfx1250) — matrix operand order repaired and checked by reading, not compiled, not run on hardware
//
#include <hip/hip_runtime.h>
#include <hip/hip_bf16.h>
#include <math.h>

// ---------------- problem constants ----------------
#define D_    256
#define H_    8
#define HD_   32
#define FF_   2048
#define EDIM_ 16
#define B_    2
#define N_    16384
#define E_    262144
#define M_    (B_ * N_)      // 32768 rows of activations

typedef __attribute__((ext_vector_type(16))) __bf16        v16bf;
typedef __attribute__((ext_vector_type(8)))  float         v8f;
typedef __attribute__((ext_vector_type(4)))  unsigned int  v4u;

// TDM descriptor operand types (clang-23 6-arg builtin form:
//  (uint32x4 g0, int32x8 g1, int32x4 g2, int32x4 g3, int32x8 g4, i32 cpol))
typedef __attribute__((ext_vector_type(4))) unsigned int tdm_g0_t;
typedef __attribute__((ext_vector_type(8))) int          tdm_g1_t;
typedef __attribute__((ext_vector_type(4))) int          tdm_g23_t;
typedef __attribute__((ext_vector_type(8))) int          tdm_g4_t;

using bf16 = __hip_bfloat16;

// ---------------- helpers ----------------
__device__ __forceinline__ unsigned encf(float f) {
    unsigned u = __float_as_uint(f);
    return (u & 0x80000000u) ? ~u : (u | 0x80000000u);
}
__device__ __forceinline__ float decf(unsigned e) {
    return __uint_as_float((e & 0x80000000u) ? (e ^ 0x80000000u) : ~e);
}

// ---------------- pack: f32 -> bf16 flat ----------------
__global__ void k_f32_to_bf16(const float* __restrict__ src,
                              bf16* __restrict__ dst, int n) {
    int i = blockIdx.x * blockDim.x + threadIdx.x;
    if (i < n) dst[i] = __float2bfloat16(src[i]);
}

// ---------------- pack fused QKV weights [K=256][N=768] + bias ----------------
__global__ void k_pack_qkv(const float* __restrict__ Wq, const float* __restrict__ Wk,
                           const float* __restrict__ Wv,
                           const float* __restrict__ bq, const float* __restrict__ bk,
                           const float* __restrict__ bv,
                           bf16* __restrict__ Wf, float* __restrict__ bqkv) {
    int i = blockIdx.x * blockDim.x + threadIdx.x;
    if (i < D_ * 3 * D_) {
        int k = i / (3 * D_), n = i % (3 * D_);
        const float* W = (n < D_) ? Wq : ((n < 2 * D_) ? Wk : Wv);
        Wf[i] = __float2bfloat16(W[k * D_ + (n & (D_ - 1))]);
    }
    if (i < 3 * D_) {
        const float* bs = (i < D_) ? bq : ((i < 2 * D_) ? bk : bv);
        bqkv[i] = bs[i & (D_ - 1)];
    }
}

// ---------------- WMMA GEMM: C[M,N] = A[M,K](bf16,row) * Bw[K,N](bf16,row) + bias ----
// Block: 256 threads = 8 waves. Block tile 128(M) x 64(N); wave tile 16 x 64.
// B tile (32x64 bf16, padded to 72-elem rows) staged in LDS by the Tensor Data
// Mover, double-buffered: wave 0 issues the DMA for tile i+1, waits TENSORcnt<=1
// (tile i done, issued one iteration earlier), then a barrier releases the block.
// ACT: 0 = none, 1 = exact GELU.  OUT_BF16: store bf16 (else f32).
template<int ACT, int OUT_BF16>
__global__ __launch_bounds__(256)
void k_gemm_wmma(const bf16* __restrict__ A, const bf16* __restrict__ Bw,
                 const float* __restrict__ bias,
                 float* __restrict__ Cf, bf16* __restrict__ Cb,
                 int M, int K, int N) {
    constexpr int LDSB = 72;                       // padded row stride (bf16 elems)
    constexpr int BUFB = 32 * LDSB * 2;            // bytes per buffer (4608)
    __shared__ __align__(16) bf16 sB[2][32 * LDSB];

    const int tid  = threadIdx.x;
    const int wave = tid >> 5;
    const int lane = tid & 31;
    const int m0   = blockIdx.y * 128 + wave * 16;
    const int n0   = blockIdx.x * 64;
    const int row  = lane & 15;
    const int hi   = lane >> 4;

    v8f zero = {0.f, 0.f, 0.f, 0.f, 0.f, 0.f, 0.f, 0.f};
    v8f acc[4] = {zero, zero, zero, zero};

    // ---- loop-invariant TDM descriptor group 1 ----
    // word0: workgroup_mask=0, data_size=1(2B), pad_enable=1,
    //        pad_interval=4 (32 DWORDs = 64 bf16 row), pad_amount=3 (4 DWORDs = 8 elems)
    const int g1w0 = (1 << 16) | (1 << 20) | (4 << 22) | (3 << 25);
    const tdm_g1_t g1 = {
        g1w0,
        (int)((unsigned)(N & 0xFFFF) << 16),               // tensor_dim0[15:0]
        (int)(((unsigned)N >> 16) | ((unsigned)(K & 0xFFFF) << 16)), // dim0 hi | dim1 lo
        (int)(((unsigned)K >> 16) | (64u << 16)),          // dim1 hi | tile_dim0=64
        32,                                                // tile_dim1=32, tile_dim2=0
        N,                                                 // tensor_dim0_stride lo32
        0, 0                                               // stride hi, dim1_stride (2D)
    };
    const tdm_g23_t gz = {0, 0, 0, 0};
    const tdm_g4_t  gz8 = {0, 0, 0, 0, 0, 0, 0, 0};
    const unsigned ldsbase = (unsigned)(size_t)(&sB[0][0]);

    const int nsteps = K / 32;

    // issue DMA for tile starting at K-row k0 into buffer `buf`
    auto tdm_issue = [&](int k0, int buf) {
        unsigned long long ga =
            (unsigned long long)(size_t)(Bw + (size_t)k0 * N + n0);
        tdm_g0_t g0;
        g0.x = 1u;                                          // count=1 (valid user D#)
        g0.y = ldsbase + (unsigned)buf * BUFB;              // lds_addr
        g0.z = (unsigned)(ga & 0xFFFFFFFFull);              // global_addr lo
        g0.w = (unsigned)((ga >> 32) & 0x1FFFFFFull) | 0x80000000u; // hi | type=2
        __builtin_amdgcn_tensor_load_to_lds(g0, g1, gz, gz, gz8, 0);
    };

    if (wave == 0) tdm_issue(0, 0);

    for (int i = 0; i < nsteps; ++i) {
        __syncthreads();   // all waves done reading buffer (i+1)&1 (iter i-1)
        if (wave == 0) {
            if (i + 1 < nsteps) {
                tdm_issue((i + 1) * 32, (i + 1) & 1);
                __builtin_amdgcn_s_wait_tensorcnt(1);       // tile i landed
            } else {
                __builtin_amdgcn_s_wait_tensorcnt(0);
            }
        }
        __syncthreads();   // buffer i&1 visible to all waves
        const bf16* sbuf = &sB[i & 1][0];
        const int   k0   = i * 32;

        // A fragment (16x32): lane<16 holds K {0..7,16..23}; lane>=16 holds {8..15,24..31}
        union { v4u u[2]; v16bf v; } af;
        const bf16* ap = A + (size_t)(m0 + row) * K + k0 + hi * 8;
        af.u[0] = *(const v4u*)(ap);
        af.u[1] = *(const v4u*)(ap + 16);

#pragma unroll
        for (int t = 0; t < 4; ++t) {
            // B fragment (32x16): lane%16 = N col; lane<16 holds K {0..7,16..23}, lane>=16 holds {8..15,24..31}
            union { v4u u[2]; v16bf v; } bfv;
            const bf16* bp = sbuf + t * 16 + row;
#pragma unroll
            for (int s = 0; s < 16; ++s)
                bfv.v[s] = *(const __bf16*)(bp + ((s & 7) + 8 * hi + 16 * (s >> 3)) * LDSB);
            acc[t] = __builtin_amdgcn_wmma_f32_16x16x32_bf16(
                false, af.v, false, bfv.v, (short)0, acc[t], false, false);
        }
    }

#pragma unroll
    for (int t = 0; t < 4; ++t) {
        const int col = n0 + t * 16 + row;
        const float bs = bias[col];
#pragma unroll
        for (int r = 0; r < 8; ++r) {
            const int orow = m0 + hi * 8 + r;
            float v = acc[t][r] + bs;
            if (ACT == 1) v = 0.5f * v * (1.0f + erff(v * 0.70710678118654752f));
            const size_t oi = (size_t)orow * N + col;
            if (OUT_BF16) Cb[oi] = __float2bfloat16(v);
            else          Cf[oi] = v;
        }
    }
}

// ---------------- per-edge scores + segment max (encoded atomicMax) ------------
__global__ void k_scores(const int* __restrict__ src, const int* __restrict__ dst,
                         const float* __restrict__ eattr, const float* __restrict__ We,
                         const float* __restrict__ be,
                         const bf16* __restrict__ qkv,
                         float* __restrict__ S, unsigned* __restrict__ menc) {
    int idx = blockIdx.x * blockDim.x + threadIdx.x;
    if (idx >= E_ * B_ * H_) return;
    const int e = idx >> 4, bh = idx & 15, b = bh >> 3, h = bh & 7;
    const int s = src[e], d = dst[e];
    const bf16* q = qkv + ((size_t)b * N_ + s) * (3 * D_) + h * HD_;
    const bf16* k = qkv + ((size_t)b * N_ + d) * (3 * D_) + D_ + h * HD_;
    float acc = 0.f;
#pragma unroll
    for (int i = 0; i < HD_; ++i)
        acc += __bfloat162float(q[i]) * __bfloat162float(k[i]);
    float bias = be[h];
#pragma unroll
    for (int i = 0; i < EDIM_; ++i)
        bias += eattr[e * EDIM_ + i] * We[i * H_ + h];
    const float sc = acc * 0.17677669529663687f + bias;   // 1/sqrt(32)
    S[idx] = sc;
    atomicMax(&menc[((size_t)s * B_ + b) * H_ + h], encf(sc));
}

// ---------------- exp(s - m[src]) + segment sum ----------------
__global__ void k_expsum(const int* __restrict__ src, float* __restrict__ S,
                         const unsigned* __restrict__ menc, float* __restrict__ z) {
    int idx = blockIdx.x * blockDim.x + threadIdx.x;
    if (idx >= E_ * B_ * H_) return;
    const int e = idx >> 4, bh = idx & 15, b = bh >> 3, h = bh & 7;
    const int s = src[e];
    const float mm = decf(menc[((size_t)s * B_ + b) * H_ + h]);
    const float ex = expf(S[idx] - mm);
    S[idx] = ex;
    atomicAdd(&z[((size_t)s * B_ + b) * H_ + h], ex);
}

// ---------------- weighted scatter-add of V[dst] into agg[src] ----------------
__global__ void k_agg(const int* __restrict__ src, const int* __restrict__ dst,
                      const float* __restrict__ S, const float* __restrict__ z,
                      const bf16* __restrict__ qkv, float* __restrict__ agg) {
    int idx = blockIdx.x * blockDim.x + threadIdx.x;
    if (idx >= E_ * B_ * H_) return;
    const int e = idx >> 4, bh = idx & 15, b = bh >> 3, h = bh & 7;
    const int s = src[e], d = dst[e];
    const float w = S[idx] / z[((size_t)s * B_ + b) * H_ + h];
    const bf16* v  = qkv + ((size_t)b * N_ + d) * (3 * D_) + 2 * D_ + h * HD_;
    float*      ag = agg + ((size_t)b * N_ + s) * D_ + h * HD_;
#pragma unroll
    for (int i = 0; i < HD_; ++i)
        atomicAdd(&ag[i], w * __bfloat162float(v[i]));
}

// ---------------- LayerNorm(X + Y)*g + b ; one wave per 256-wide row ----------
template<int WRITE_BF16>
__global__ __launch_bounds__(256)
void k_ln(const float* __restrict__ X, const float* __restrict__ Y,
          const float* __restrict__ g, const float* __restrict__ bt,
          float* __restrict__ Of, bf16* __restrict__ Ob) {
    const int wave = threadIdx.x >> 5, lane = threadIdx.x & 31;
    const size_t rowid = (size_t)blockIdx.x * 8 + wave;
    const float* x = X + rowid * D_;
    const float* y = Y + rowid * D_;
    float v[8];
    float s = 0.f, ss = 0.f;
#pragma unroll
    for (int j = 0; j < 2; ++j) {
        float4 a = *(const float4*)(x + lane * 8 + j * 4);
        float4 c = *(const float4*)(y + lane * 8 + j * 4);
        v[j * 4 + 0] = a.x + c.x; v[j * 4 + 1] = a.y + c.y;
        v[j * 4 + 2] = a.z + c.z; v[j * 4 + 3] = a.w + c.w;
    }
#pragma unroll
    for (int i = 0; i < 8; ++i) { s += v[i]; ss += v[i] * v[i]; }
#pragma unroll
    for (int m = 16; m >= 1; m >>= 1) {
        s  += __shfl_xor(s,  m, 32);
        ss += __shfl_xor(ss, m, 32);
    }
    const float mean = s * (1.f / 256.f);
    const float var  = ss * (1.f / 256.f) - mean * mean;
    const float rstd = rsqrtf(var + 1e-5f);
#pragma unroll
    for (int i = 0; i < 8; ++i) {
        const int c = lane * 8 + i;
        const float o = (v[i] - mean) * rstd * g[c] + bt[c];
        Of[rowid * D_ + c] = o;
        if (WRITE_BF16) Ob[rowid * D_ + c] = __float2bfloat16(o);
    }
}

// ---------------- host-side launch ----------------
extern "C" void kernel_launch(void* const* d_in, const int* in_sizes, int n_in,
                              void* d_out, int out_size, void* d_ws, size_t ws_size,
                              hipStream_t stream) {
    const float* x     = (const float*)d_in[0];
    const int*   ei    = (const int*)d_in[1];
    const float* eattr = (const float*)d_in[2];
    const float* Wq    = (const float*)d_in[3];
    const float* bq    = (const float*)d_in[4];
    const float* Wk    = (const float*)d_in[5];
    const float* bk    = (const float*)d_in[6];
    const float* Wv    = (const float*)d_in[7];
    const float* bv    = (const float*)d_in[8];
    const float* We    = (const float*)d_in[9];
    const float* be    = (const float*)d_in[10];
    const float* Wo    = (const float*)d_in[11];
    const float* bo    = (const float*)d_in[12];
    const float* g1    = (const float*)d_in[13];
    const float* be1   = (const float*)d_in[14];
    const float* W1    = (const float*)d_in[15];
    const float* bf1   = (const float*)d_in[16];
    const float* W2    = (const float*)d_in[17];
    const float* bf2   = (const float*)d_in[18];
    const float* g2    = (const float*)d_in[19];
    const float* be2   = (const float*)d_in[20];
    float* out = (float*)d_out;

    const int* srcI = ei;        // edge_index[0]
    const int* dstI = ei + E_;   // edge_index[1]

    // ---- workspace carve-up ----
    char* p = (char*)d_ws;
    auto take = [&](size_t bytes) -> void* {
        void* r = (void*)p;
        p += (bytes + 255) & ~(size_t)255;
        return r;
    };
    bf16*     xb    = (bf16*)take((size_t)M_ * D_ * 2);
    bf16*     Wfq   = (bf16*)take((size_t)D_ * 3 * D_ * 2);
    float*    bqkv  = (float*)take((size_t)3 * D_ * 4);
    bf16*     Wob   = (bf16*)take((size_t)D_ * D_ * 2);
    bf16*     W1b   = (bf16*)take((size_t)D_ * FF_ * 2);
    bf16*     W2b   = (bf16*)take((size_t)FF_ * D_ * 2);
    bf16*     qkvb  = (bf16*)take((size_t)M_ * 3 * D_ * 2);
    float*    S     = (float*)take((size_t)E_ * B_ * H_ * 4);
    unsigned* menc  = (unsigned*)take((size_t)N_ * B_ * H_ * 4);
    float*    z     = (float*)take((size_t)N_ * B_ * H_ * 4);
    float*    agg   = (float*)take((size_t)M_ * D_ * 4);
    bf16*     aggb  = (bf16*)take((size_t)M_ * D_ * 2);
    float*    attn  = (float*)take((size_t)M_ * D_ * 4);
    float*    h     = (float*)take((size_t)M_ * D_ * 4);
    bf16*     hb    = (bf16*)take((size_t)M_ * D_ * 2);
    bf16*     gout  = (bf16*)take((size_t)M_ * FF_ * 2);
    float*    ffn   = (float*)take((size_t)M_ * D_ * 4);

    // ---- zero segment state ----
    (void)hipMemsetAsync(menc, 0, (size_t)N_ * B_ * H_ * 4, stream); // enc(-inf) sentinel
    (void)hipMemsetAsync(z,    0, (size_t)N_ * B_ * H_ * 4, stream);
    (void)hipMemsetAsync(agg,  0, (size_t)M_ * D_ * 4, stream);

    // ---- pack activations / weights to bf16 ----
    {
        int n = M_ * D_;
        k_f32_to_bf16<<<(n + 255) / 256, 256, 0, stream>>>(x, xb, n);
    }
    {
        int n = D_ * 3 * D_;
        k_pack_qkv<<<(n + 255) / 256, 256, 0, stream>>>(Wq, Wk, Wv, bq, bk, bv, Wfq, bqkv);
    }
    { int n = D_ * D_;  k_f32_to_bf16<<<(n + 255) / 256, 256, 0, stream>>>(Wo, Wob, n); }
    { int n = D_ * FF_; k_f32_to_bf16<<<(n + 255) / 256, 256, 0, stream>>>(W1, W1b, n); }
    { int n = FF_ * D_; k_f32_to_bf16<<<(n + 255) / 256, 256, 0, stream>>>(W2, W2b, n); }

    // ---- fused QKV projection: [32768,256] x [256,768] -> bf16 ----
    {
        dim3 grid((3 * D_) / 64, M_ / 128);
        k_gemm_wmma<0, 1><<<grid, 256, 0, stream>>>(xb, Wfq, bqkv, nullptr, qkvb,
                                                    M_, D_, 3 * D_);
    }

    // ---- edge attention pipeline ----
    {
        int n = E_ * B_ * H_;
        k_scores<<<(n + 255) / 256, 256, 0, stream>>>(srcI, dstI, eattr, We, be,
                                                      qkvb, S, menc);
        k_expsum<<<(n + 255) / 256, 256, 0, stream>>>(srcI, S, menc, z);
        k_agg<<<(n + 255) / 256, 256, 0, stream>>>(srcI, dstI, S, z, qkvb, agg);
    }

    // ---- output projection: agg @ Wo + bo ----
    {
        int n = M_ * D_;
        k_f32_to_bf16<<<(n + 255) / 256, 256, 0, stream>>>(agg, aggb, n);
        dim3 grid(D_ / 64, M_ / 128);
        k_gemm_wmma<0, 0><<<grid, 256, 0, stream>>>(aggb, Wob, bo, attn, nullptr,
                                                    M_, D_, D_);
    }

    // ---- LN1: h = LN(x + attn) ----
    k_ln<1><<<M_ / 8, 256, 0, stream>>>(x, attn, g1, be1, h, hb);

    // ---- FFN1: gelu(h @ W1 + bf1) -> bf16 ----
    {
        dim3 grid(FF_ / 64, M_ / 128);
        k_gemm_wmma<1, 1><<<grid, 256, 0, stream>>>(hb, W1b, bf1, nullptr, gout,
                                                    M_, D_, FF_);
    }
    // ---- FFN2: gout @ W2 + bf2 -> f32 ----
    {
        dim3 grid(D_ / 64, M_ / 128);
        k_gemm_wmma<0, 0><<<grid, 256, 0, stream>>>(gout, W2b, bf2, ffn, nullptr,
                                                    M_, FF_, D_);
    }

    // ---- LN2: out = LN(h + ffn) ----
    k_ln<0><<<M_ / 8, 256, 0, stream>>>(h, ffn, g2, be2, out, nullptr);
}
